// TanhNewtonLayer_38551626449607
// MI455X (gfx1250) — matrix operand order repaired and checked by reading, not compiled, not run on hardware
//
#include <hip/hip_runtime.h>
#include <hip/hip_bf16.h>
#include <math.h>

// ---------------------------------------------------------------------------
// Newton solve of z = tanh(W z + b + x), B=256, N=384, fp32.
//   * residual GEMM (zl = z@W^T + b + x) in fp32 WMMA  (precision-critical)
//   * Newton direction dz via truncated Richardson  dz <- g + sech2 (.) (dz@W^T)
//     FUSED: one workgroup owns a 16-row slab, dz lives in LDS, all 24 inner
//     steps run inside one kernel on bf16 WMMA (launch-latency was the
//     bottleneck -- each inner GEMM is only 75 MFLOP, everything L2-resident).
// ---------------------------------------------------------------------------

#define BATCH 256
#define NDIM  384
#define NELEM (BATCH * NDIM)          // 98304
#define TILES_M (BATCH / 16)          // 16
#define TILES_N (NDIM / 16)           // 24
#define NTILES  (TILES_M * TILES_N)   // 384
#define WAVES_PER_BLOCK 8
#define GEMM_BLOCKS (NTILES / WAVES_PER_BLOCK) // 48
#define NEWTON_ITERS 10
#define INNER_STEPS  24
#define SLAB  (16 * NDIM)             // 6144 elements per 16-row slab
#define TOLF 1e-4f

typedef __attribute__((ext_vector_type(2)))  float   v2f;
typedef __attribute__((ext_vector_type(8)))  float   v8f;
typedef __attribute__((ext_vector_type(8)))  __bf16  v8bf;
typedef __attribute__((ext_vector_type(16))) __bf16  v16bf;

// ---------------------------------------------------------------------------
// fp32 WMMA main loop: C(16x16) += A(16x4) * B(4x16) over K = 384.
// A-frag (32-bit A 16x4 layout): lane l holds M = l&15, VGPR v <-> K = v + 2*(l>>4).
// B-frag mirrors: lane l holds N = l&15, VGPR v <-> K = v + 2*(l>>4).
// B[k][n] = W[n][k]  (row-major W, read along k -> contiguous float2).
// ---------------------------------------------------------------------------
__device__ inline v8f wmma_tile_f32(const float* __restrict__ A,
                                    const float* __restrict__ W,
                                    int tileM, int tileN, int lane) {
  const int hsel = lane >> 4;
  const int l16  = lane & 15;
  v8f c = {};
#if __has_builtin(__builtin_amdgcn_wmma_f32_16x16x4_f32)
  const float* arow = A + (tileM + l16) * NDIM + 2 * hsel;
  const float* brow = W + (tileN + l16) * NDIM + 2 * hsel;
#pragma unroll 8
  for (int k = 0; k < NDIM; k += 4) {
    v2f a = *(const v2f*)(arow + k);
    v2f b = *(const v2f*)(brow + k);
    c = __builtin_amdgcn_wmma_f32_16x16x4_f32(false, a, false, b,
                                              (short)0, c, false, false);
  }
#else
  const int n = tileN + l16;
#pragma unroll
  for (int r = 0; r < 8; ++r) {
    const int m = tileM + r + 8 * hsel;
    float acc = 0.f;
    for (int k = 0; k < NDIM; ++k)
      acc = fmaf(A[m * NDIM + k], W[n * NDIM + k], acc);
    c[r] = acc;
  }
#endif
  return c;
}

// bf16 A-fragment builder (contiguous k order, matching the B loads): two 16B loads at
// k0+16*hsel and k0+16*hsel+8 from row (l&15).  Works for LDS or global ptr.
__device__ inline v16bf bf16_afrag(const __bf16* arow, int k0, int hsel) {
  v8bf alo = *(const v8bf*)(arow + k0 + 16 * hsel);
  v8bf ahi = *(const v8bf*)(arow + k0 + 16 * hsel + 8);
  return __builtin_shufflevector(alo, ahi,
      0, 1, 2, 3, 4, 5, 6, 7, 8, 9, 10, 11, 12, 13, 14, 15);
}

// ---------------------------------------------------------------------------
// Kernels
// ---------------------------------------------------------------------------
__global__ void init_kernel(const float* __restrict__ x, float* __restrict__ z,
                            float* __restrict__ errsq, float* __restrict__ done) {
  int i = blockIdx.x * blockDim.x + threadIdx.x;
  if (i < NELEM) z[i] = tanhf(x[i]);
  if (i == 0) { errsq[0] = 0.f; done[0] = 0.f; }
}

__global__ void convert_w_kernel(const float* __restrict__ W,
                                 __bf16* __restrict__ Wbf) {
  int i = blockIdx.x * blockDim.x + threadIdx.x;
  if (i < NDIM * NDIM) Wbf[i] = (__bf16)W[i];
}

// zl = z@W^T + b + x ; g = z - tanh(zl) ; sech2 = 1 - tanh^2 ; ||g||^2 -> errsq
__global__ __launch_bounds__(256)
void newton_residual_kernel(const float* __restrict__ z,
                            const float* __restrict__ W,
                            const float* __restrict__ x,
                            const float* __restrict__ bias,
                            float* __restrict__ g,
                            float* __restrict__ sech2,
                            float* __restrict__ errsq) {
  const int lane = threadIdx.x & 31;
  const int wave = threadIdx.x >> 5;
  const int tile = blockIdx.x * WAVES_PER_BLOCK + wave;
  const int tileM = (tile & (TILES_M - 1)) * 16;
  const int tileN = (tile / TILES_M) * 16;

  v8f c = wmma_tile_f32(z, W, tileM, tileN, lane);

  const int hsel = lane >> 4;
  const int l16  = lane & 15;
  const int n = tileN + l16;
  const float bn = bias[n];
  float esum = 0.f;
#pragma unroll
  for (int r = 0; r < 8; ++r) {
    const int m = tileM + r + 8 * hsel;
    const int idx = m * NDIM + n;
    const float zl = c[r] + bn + x[idx];
    const float t = tanhf(zl);
    const float gv = z[idx] - t;
    g[idx] = gv;
    sech2[idx] = 1.0f - t * t;
    esum += gv * gv;
  }
#pragma unroll
  for (int off = 16; off > 0; off >>= 1) esum += __shfl_xor(esum, off, 32);
  if (lane == 0) atomicAdd(errsq, esum);
}

__global__ void flag_kernel(float* __restrict__ errsq, float* __restrict__ done) {
  if (blockIdx.x == 0 && threadIdx.x == 0) {
    if (sqrtf(errsq[0]) <= TOLF) done[0] = 1.0f; // sticky, as in reference
    errsq[0] = 0.f;                              // rearm for next iteration
  }
}

// ---------------------------------------------------------------------------
// Fused Richardson solver + z update.  Grid: 16 blocks (one 16-row slab each),
// 384 threads = 12 waves, each wave owns 2 of the 24 N-tiles.  dz ping-pongs
// in LDS (bf16); g (f32) and sech2 (bf16) staged in LDS.  60 KB LDS total.
//   per step: dz_nxt = g + sech2 (.) (dz_cur @ W^T)   [bf16 WMMA, K=384]
// After INNER_STEPS steps: z -= dz  (skipped entirely when done-flag set).
// ---------------------------------------------------------------------------
__global__ __launch_bounds__(384)
void newton_inner_fused_kernel(const float* __restrict__ g,
                               const float* __restrict__ sech2,
                               const __bf16* __restrict__ Wbf,
                               float* __restrict__ z,
                               const float* __restrict__ done) {
  __shared__ __bf16 dzs[2][SLAB];   // 2 x 12 KB
  __shared__ float  gs[SLAB];       // 24 KB
  __shared__ __bf16 ss[SLAB];       // 12 KB

  if (done[0] != 0.0f) return;      // uniform across all threads

  const int m0 = blockIdx.x * 16;
  for (int i = threadIdx.x; i < SLAB; i += 384) {
    const float gv = g[m0 * NDIM + i];
    gs[i] = gv;
    ss[i] = (__bf16)sech2[m0 * NDIM + i];
    dzs[0][i] = (__bf16)gv;         // dz_0 = g
  }
  __syncthreads();

  const int lane = threadIdx.x & 31;
  const int wave = threadIdx.x >> 5;          // 0..11
  const int hsel = lane >> 4;
  const int l16  = lane & 15;
  const int n0 = (wave * 2 + 0) * 16 + l16;   // tile 0 column
  const int n1 = (wave * 2 + 1) * 16 + l16;   // tile 1 column
  const __bf16* b0row = Wbf + n0 * NDIM;
  const __bf16* b1row = Wbf + n1 * NDIM;

  int cur = 0;
  for (int step = 0; step < INNER_STEPS; ++step) {
    const __bf16* arow = &dzs[cur][l16 * NDIM];
    v8f c0 = {}, c1 = {};
#pragma unroll 4
    for (int k0 = 0; k0 < NDIM; k0 += 32) {
      v16bf a  = bf16_afrag(arow, k0, hsel);            // LDS, shared by waves
      v16bf b0 = *(const v16bf*)(b0row + k0 + 16 * hsel); // L2-resident W
      v16bf b1 = *(const v16bf*)(b1row + k0 + 16 * hsel);
      c0 = __builtin_amdgcn_wmma_f32_16x16x32_bf16(false, a, false, b0,
                                                   (short)0, c0, false, false);
      c1 = __builtin_amdgcn_wmma_f32_16x16x32_bf16(false, a, false, b1,
                                                   (short)0, c1, false, false);
    }
    const int nxt = cur ^ 1;
#pragma unroll
    for (int r = 0; r < 8; ++r) {
      const int ml = (r + 8 * hsel) * NDIM;
      const int i0 = ml + n0;
      const int i1 = ml + n1;
      dzs[nxt][i0] = (__bf16)(gs[i0] + (float)ss[i0] * c0[r]);
      dzs[nxt][i1] = (__bf16)(gs[i1] + (float)ss[i1] * c1[r]);
    }
    __syncthreads();                 // writes visible before next step's reads
    cur = nxt;
  }

  for (int i = threadIdx.x; i < SLAB; i += 384)
    z[m0 * NDIM + i] -= (float)dzs[cur][i];
}

// Per-row ||g|| check; zero failed rows; write output.
__global__ __launch_bounds__(128)
void finalize_kernel(const float* __restrict__ z, const float* __restrict__ g,
                     float* __restrict__ out) {
  __shared__ float red[128];
  const int row = blockIdx.x;
  float s = 0.f;
  for (int j = threadIdx.x; j < NDIM; j += blockDim.x) {
    float gv = g[row * NDIM + j];
    s += gv * gv;
  }
  red[threadIdx.x] = s;
  __syncthreads();
  for (int o = 64; o > 0; o >>= 1) {
    if (threadIdx.x < o) red[threadIdx.x] += red[threadIdx.x + o];
    __syncthreads();
  }
  const float keep = (sqrtf(red[0]) > TOLF) ? 0.f : 1.f;
  for (int j = threadIdx.x; j < NDIM; j += blockDim.x)
    out[row * NDIM + j] = keep * z[row * NDIM + j];
}

// ---------------------------------------------------------------------------
// Host launch: fixed, graph-capturable schedule (34 launches total).
// ---------------------------------------------------------------------------
extern "C" void kernel_launch(void* const* d_in, const int* in_sizes, int n_in,
                              void* d_out, int out_size, void* d_ws, size_t ws_size,
                              hipStream_t stream) {
  const float* x    = (const float*)d_in[0];
  const float* W    = (const float*)d_in[1];
  const float* bias = (const float*)d_in[2];
  float* out = (float*)d_out;

  char* ws = (char*)d_ws;
  size_t off = 0;
  auto carve = [&](size_t bytes) {
    char* p = ws + off;
    off += (bytes + 255) & ~(size_t)255;
    return p;
  };
  float*  z     = (float*)carve(NELEM * sizeof(float));
  float*  g     = (float*)carve(NELEM * sizeof(float));
  float*  sech2 = (float*)carve(NELEM * sizeof(float));
  __bf16* Wbf   = (__bf16*)carve(NDIM * NDIM * sizeof(__bf16));
  float*  errsq = (float*)carve(2 * sizeof(float));
  float*  done  = errsq + 1;

  init_kernel<<<(NELEM + 255) / 256, 256, 0, stream>>>(x, z, errsq, done);
  convert_w_kernel<<<(NDIM * NDIM + 255) / 256, 256, 0, stream>>>(W, Wbf);

  for (int it = 0; it < NEWTON_ITERS; ++it) {
    newton_residual_kernel<<<GEMM_BLOCKS, 256, 0, stream>>>(
        z, W, x, bias, g, sech2, errsq);
    flag_kernel<<<1, 32, 0, stream>>>(errsq, done);
    newton_inner_fused_kernel<<<TILES_M, 384, 0, stream>>>(
        g, sech2, Wbf, z, done);
  }

  // final residual for the per-row convergence mask
  newton_residual_kernel<<<GEMM_BLOCKS, 256, 0, stream>>>(
      z, W, x, bias, g, sech2, errsq);
  finalize_kernel<<<BATCH, 128, 0, stream>>>(z, g, out);
}
